// DentalMetricDGCNN_3504693313576
// MI455X (gfx1250) — matrix operand order repaired and checked by reading, not compiled, not run on hardware
//
#include <hip/hip_runtime.h>
#include <hip/hip_bf16.h>

// ---------------------------------------------------------------------------
// DGCNN forward for MI455X (gfx1250, wave32, WMMA + TDM).
// Input order (jax tree flatten, dicts sorted by key inside params):
//   d_in[0] = pos   [8*4096*3] f32
//   d_in[1] = label [32768]    i32
//   d_in[2] = arc_W [3*128]    f32
//   then groups of 4 (W,b,be,g) for: c1a,c1b,c2a,c2b,c3a,c3b,e1,e2,e3,g1,g2
// Output: [32768, 3] f32 arcface logits.
// ---------------------------------------------------------------------------

typedef _Float16 f16;
typedef _Float16 v16h __attribute__((ext_vector_type(16)));
typedef _Float16 v8h  __attribute__((ext_vector_type(8)));
typedef float    v8f  __attribute__((ext_vector_type(8)));
typedef unsigned int u32x4 __attribute__((ext_vector_type(4)));
typedef int          i32x8 __attribute__((ext_vector_type(8)));
typedef int          i32x4 __attribute__((ext_vector_type(4)));

#define BATCH   8
#define NPTS    4096
#define KNN     20
#define BN_ROWS (BATCH * NPTS)       /* 32768  */
#define ME_ROWS (BN_ROWS * KNN)      /* 655360 */
#define BN_EPS  1e-5f
#define S_ARC   30.0f
#define M_ARC   0.5f
#define GEMM_KC 128                  /* K-chunk staged in LDS */

// ---------------------------- utility kernels ------------------------------

__global__ void k_zero(float* p, int n) {
  int t = blockIdx.x * blockDim.x + threadIdx.x;
  if (t < n) p[t] = 0.f;
}

// f32 weight [Nout,Kd] -> f16 padded [Nout,Kp] (zero pad K to mult of 32)
__global__ void k_w2h(const float* __restrict__ W, f16* __restrict__ Wh,
                      int Nout, int Kd, int Kp) {
  int t = blockIdx.x * blockDim.x + threadIdx.x;
  if (t >= Nout * Kp) return;
  int n = t / Kp, c = t % Kp;
  Wh[t] = (c < Kd) ? (f16)W[n * Kd + c] : (f16)0.f;
}

// ------------------------------- kNN ---------------------------------------
// One wave per query point. Per-lane sorted top-20 list kept in LDS,
// then a 32-lane merge via wave-argmin shuffles. Self (dist 0) included,
// matching jax.lax.top_k(-dist, K).
__global__ __launch_bounds__(256) void k_knn(const float* __restrict__ x,
                                             int d, int* __restrict__ idx) {
  __shared__ float s_xi[8][128];
  __shared__ float s_d[8][32][KNN];
  __shared__ int   s_j[8][32][KNN];

  int wave = threadIdx.x >> 5;
  int lane = threadIdx.x & 31;
  int row  = blockIdx.x * 8 + wave;          // global point id
  int b    = row >> 12;                      // / 4096
  int i    = row & 4095;
  const float* xb = x + (size_t)b * NPTS * d;

  for (int c = lane; c < d; c += 32) s_xi[wave][c] = xb[(size_t)i * d + c];
  __syncthreads();

  float* ld = &s_d[wave][lane][0];
  int*   lj = &s_j[wave][lane][0];
  for (int t = 0; t < KNN; ++t) { ld[t] = 3.4e38f; lj[t] = 0; }

  for (int j = lane; j < NPTS; j += 32) {
    const float* xj = xb + (size_t)j * d;
    float dist = 0.f;
    for (int c = 0; c < d; ++c) {
      float df = s_xi[wave][c] - xj[c];
      dist = fmaf(df, df, dist);
    }
    if (dist < ld[KNN - 1]) {
      int p = KNN - 1;
      while (p > 0 && ld[p - 1] > dist) {
        ld[p] = ld[p - 1]; lj[p] = lj[p - 1]; --p;
      }
      ld[p] = dist; lj[p] = j;
    }
  }

  // merge 32 sorted lists -> first KNN
  int ptr = 0;
  for (int s = 0; s < KNN; ++s) {
    float v  = (ptr < KNN) ? ld[ptr] : 3.4e38f;
    int   jv = (ptr < KNN) ? lj[ptr] : 0;
    int   wl = lane;
    for (int off = 16; off >= 1; off >>= 1) {
      float ov = __shfl_xor(v, off, 32);
      int   oj = __shfl_xor(jv, off, 32);
      int   ol = __shfl_xor(wl, off, 32);
      if (ov < v) { v = ov; jv = oj; wl = ol; }
    }
    if (lane == 0) idx[(size_t)row * KNN + s] = jv;
    if (lane == wl) ++ptr;
  }
}

// --------------------- edge feature construction (f16) ----------------------
// E[r, :] = [x_i (d), x_j - x_i (d), zeros pad] ; r = ((b*N+i)*K + k)
__global__ void k_edge_feat(const float* __restrict__ x,
                            const int* __restrict__ idx,
                            f16* __restrict__ E, int d, int Kp) {
  int t = blockIdx.x * blockDim.x + threadIdx.x;
  if (t >= ME_ROWS * Kp) return;
  int r = t / Kp, c = t % Kp;
  int b = r / (NPTS * KNN);
  int i = (r / KNN) & 4095;
  const float* xb = x + (size_t)b * NPTS * d;
  float val = 0.f;
  if (c < d) {
    val = xb[(size_t)i * d + c];
  } else if (c < 2 * d) {
    int cc = c - d;
    int j  = idx[r];
    val = xb[(size_t)j * d + cc] - xb[(size_t)i * d + cc];
  }
  E[(size_t)t] = (f16)val;
}

// ------------------------------ WMMA GEMM ----------------------------------
// Z[M,Nout] = A[M,Kp]_f16 @ W[Nout,Kp]_f16^T, fp32 accumulate.
// Block = 256 threads = 8 waves = 128(M) x 64(N) tile. The 64 x KC f16
// weight tile is DMA'd once per K-chunk into LDS by the Tensor Data Mover
// (wave 0 issues TENSOR_LOAD_TO_LDS, waits TENSORcnt, block barrier), then
// all 8 waves consume it; A streams from global with prefetch.
__global__ __launch_bounds__(256) void k_gemm(const f16* __restrict__ A,
                                              const f16* __restrict__ Bw,
                                              float* __restrict__ Z,
                                              int Kp, int Nout) {
  __shared__ f16 s_b[64 * GEMM_KC];   // 16 KB B staging
  int lane = threadIdx.x & 31;
  int wave = threadIdx.x >> 5;
  int m0   = blockIdx.x * 128 + wave * 16;
  int n0   = blockIdx.y * 64;
  int lm   = lane & 15;        // row/col within 16-tile
  int g    = lane >> 4;        // lane half

  v8f acc0 = {}, acc1 = {}, acc2 = {}, acc3 = {};

  // A tile: lane holds row (m0+lm); element e -> k = g*16 + e
  const f16* arow = A + (size_t)(m0 + lm) * Kp;
  const int chunk = (Kp < GEMM_KC) ? Kp : GEMM_KC;

  for (int kb = 0; kb < Kp; kb += chunk) {
    __syncthreads();   // previous chunk fully consumed before restaging
#if __has_builtin(__builtin_amdgcn_tensor_load_to_lds)
    if (wave == 0) {
      // Tensor DMA descriptor (D#), ISA 8.3/8.4: 2-D tile, 64 rows x chunk
      // f16 elements of W starting at row n0, col kb; row stride Kp.
      unsigned lds_off = (unsigned)(size_t)(&s_b[0]);
      unsigned long long gaddr =
          (unsigned long long)(size_t)(Bw + (size_t)n0 * Kp + kb);
      u32x4 g0;
      g0.x = 1u;                                   // count=1, user mode
      g0.y = lds_off;                              // lds_addr (bytes)
      g0.z = (unsigned)(gaddr & 0xffffffffu);      // global_addr[31:0]
      g0.w = (unsigned)((gaddr >> 32) & 0x01ffffffu) | (2u << 30); // [56:32]|type=2
      i32x8 g1;
      g1[0] = (1 << 16);                           // data_size=1 (2B), mask=0
      g1[1] = (int)(((unsigned)Kp & 0xffffu) << 16);           // tensor_dim0 lo16
      g1[2] = (int)((((unsigned)Kp >> 16) & 0xffffu)
                    | (((unsigned)Nout & 0xffffu) << 16));     // dim0 hi | dim1 lo
      g1[3] = (int)((((unsigned)Nout >> 16) & 0xffffu)
                    | ((unsigned)chunk << 16));                // dim1 hi | tile_dim0
      g1[4] = 64;                                  // tile_dim1=64, tile_dim2=0
      g1[5] = Kp;                                  // tensor_dim0_stride lo32
      g1[6] = 0;                                   // stride hi | dim1_stride lo
      g1[7] = 0;
      i32x4 gz4 = {0, 0, 0, 0};
      i32x8 gz8 = {0, 0, 0, 0, 0, 0, 0, 0};
      __builtin_amdgcn_tensor_load_to_lds(g0, g1, gz4, gz4, gz8, 0);
      __builtin_amdgcn_s_wait_tensorcnt((short)0);
    }
#else
    // cooperative staging fallback (global -> LDS via VGPRs)
    for (int t = threadIdx.x; t < 64 * (chunk / 8); t += 256) {
      int r = t / (chunk / 8), c8 = t % (chunk / 8);
      *(v8h*)&s_b[r * chunk + c8 * 8] =
          *(const v8h*)(Bw + (size_t)(n0 + r) * Kp + kb + c8 * 8);
    }
#endif
    __syncthreads();

    // B tiles from LDS: lane holds col (lm + 16*t); element e -> k = g*16 + e
    const f16* bl0 = &s_b[(size_t)lm * chunk + g * 16];
    const f16* bl1 = &s_b[(size_t)(lm + 16) * chunk + g * 16];
    const f16* bl2 = &s_b[(size_t)(lm + 32) * chunk + g * 16];
    const f16* bl3 = &s_b[(size_t)(lm + 48) * chunk + g * 16];

    for (int k0 = 0; k0 < chunk; k0 += 32) {
      __builtin_prefetch((const void*)(arow + kb + k0 + 32), 0, 1);
      v16h a = *(const v16h*)(arow + kb + k0 + g * 16);
      v16h b0 = *(const v16h*)(bl0 + k0);
      v16h b1 = *(const v16h*)(bl1 + k0);
      v16h b2 = *(const v16h*)(bl2 + k0);
      v16h b3 = *(const v16h*)(bl3 + k0);
      acc0 = __builtin_amdgcn_wmma_f32_16x16x32_f16(false, a, false, b0,
                                                    (short)0, acc0, false, false);
      acc1 = __builtin_amdgcn_wmma_f32_16x16x32_f16(false, a, false, b1,
                                                    (short)0, acc1, false, false);
      acc2 = __builtin_amdgcn_wmma_f32_16x16x32_f16(false, a, false, b2,
                                                    (short)0, acc2, false, false);
      acc3 = __builtin_amdgcn_wmma_f32_16x16x32_f16(false, a, false, b3,
                                                    (short)0, acc3, false, false);
    }
  }

  // D layout: lane n = lm, VGPR v -> m = v + 8*g
#pragma unroll
  for (int v = 0; v < 8; ++v) {
    size_t m = (size_t)(m0 + v + 8 * g);
    Z[m * Nout + n0 + lm]      = acc0[v];
    Z[m * Nout + n0 + 16 + lm] = acc1[v];
    Z[m * Nout + n0 + 32 + lm] = acc2[v];
    Z[m * Nout + n0 + 48 + lm] = acc3[v];
  }
}

// --------------------------- BN statistics ---------------------------------
// stats[0:C] = sum, stats[C:2C] = sumsq (atomic partial sums per block).
__global__ __launch_bounds__(256) void k_bn_stats(const float* __restrict__ Z,
                                                  float* __restrict__ stats,
                                                  int M, int C) {
  int c0 = threadIdx.x;
  float ps0 = 0, pq0 = 0, ps1 = 0, pq1 = 0, ps2 = 0, pq2 = 0, ps3 = 0, pq3 = 0;
  for (int r = blockIdx.x; r < M; r += gridDim.x) {
    const float* zr = Z + (size_t)r * C;
    if (c0 < C)       { float z = zr[c0];       ps0 += z; pq0 += z * z; }
    if (c0 + 256 < C) { float z = zr[c0 + 256]; ps1 += z; pq1 += z * z; }
    if (c0 + 512 < C) { float z = zr[c0 + 512]; ps2 += z; pq2 += z * z; }
    if (c0 + 768 < C) { float z = zr[c0 + 768]; ps3 += z; pq3 += z * z; }
  }
  if (c0 < C)       { atomicAdd(&stats[c0], ps0);       atomicAdd(&stats[C + c0], pq0); }
  if (c0 + 256 < C) { atomicAdd(&stats[c0 + 256], ps1); atomicAdd(&stats[C + c0 + 256], pq1); }
  if (c0 + 512 < C) { atomicAdd(&stats[c0 + 512], ps2); atomicAdd(&stats[C + c0 + 512], pq2); }
  if (c0 + 768 < C) { atomicAdd(&stats[c0 + 768], ps3); atomicAdd(&stats[C + c0 + 768], pq3); }
}

// ----------------------- BN (+ReLU) -> f16 and/or f32 -----------------------
__global__ void k_bn_act(const float* __restrict__ Z,
                         const float* __restrict__ stats,
                         const float* __restrict__ gamma,
                         const float* __restrict__ beta,
                         f16* dstH, int ldH, int colOff, float* dstF,
                         int M, int C, float invM, int relu) {
  int t = blockIdx.x * blockDim.x + threadIdx.x;
  if (t >= M * C) return;
  int r = t / C, c = t % C;
  float mean = stats[c] * invM;
  float var  = stats[C + c] * invM - mean * mean;
  float y = (Z[(size_t)t] - mean) * rsqrtf(var + BN_EPS) * gamma[c] + beta[c];
  if (relu) y = fmaxf(y, 0.f);
  if (dstH) dstH[(size_t)r * ldH + colOff + c] = (f16)y;
  if (dstF) dstF[(size_t)t] = y;
}

// ---------------- BN+ReLU fused with max-over-K aggregation -----------------
__global__ void k_max_bn(const float* __restrict__ Z2,
                         const float* __restrict__ stats,
                         const float* __restrict__ gamma,
                         const float* __restrict__ beta,
                         float* __restrict__ xout,
                         f16* __restrict__ comb, int combLd, int colOff,
                         int C, float invM) {
  int t = blockIdx.x * blockDim.x + threadIdx.x;
  if (t >= BN_ROWS * C) return;
  int p = t / C, c = t % C;
  float mean = stats[c] * invM;
  float inv  = rsqrtf(stats[C + c] * invM - mean * mean + BN_EPS);
  float gm = gamma[c], bt = beta[c];
  float m = -3.4e38f;
  const float* zp = Z2 + ((size_t)p * KNN) * C + c;
  for (int k = 0; k < KNN; ++k) {
    float y = (zp[(size_t)k * C] - mean) * inv * gm + bt;
    y = fmaxf(y, 0.f);
    m = fmaxf(m, y);
  }
  xout[(size_t)t] = m;
  comb[(size_t)p * combLd + colOff + c] = (f16)m;
}

// comb f16 -> cat1280 f16 (col copy)
__global__ void k_copy_h(const f16* __restrict__ src, f16* __restrict__ dst,
                         int srcLd, int dstLd, int cols) {
  int t = blockIdx.x * blockDim.x + threadIdx.x;
  if (t >= BN_ROWS * cols) return;
  int r = t / cols, c = t % cols;
  dst[(size_t)r * dstLd + c] = src[(size_t)r * srcLd + c];
}

// normalize arcface weight rows [3,128]
__global__ void k_wnorm(const float* __restrict__ W, float* __restrict__ wn) {
  __shared__ float red[128];
  int t = threadIdx.x;
  for (int cls = 0; cls < 3; ++cls) {
    float v = W[cls * 128 + t];
    red[t] = v * v;
    __syncthreads();
    for (int s = 64; s >= 1; s >>= 1) {
      if (t < s) red[t] += red[t + s];
      __syncthreads();
    }
    float inv = rsqrtf(red[0] + 1e-24f);
    __syncthreads();
    wn[cls * 128 + t] = v * inv;
  }
}

// ArcFace head: wave per row (8 rows per 256-thread block).
__global__ __launch_bounds__(256) void k_arcface(const float* __restrict__ emb,
                                                 const float* __restrict__ wn,
                                                 const int* __restrict__ label,
                                                 float* __restrict__ out) {
  int wave = threadIdx.x >> 5;
  int lane = threadIdx.x & 31;
  int row  = blockIdx.x * 8 + wave;
  const float* e = emb + (size_t)row * 128;
  float ev[4];
#pragma unroll
  for (int u = 0; u < 4; ++u) ev[u] = e[lane + 32 * u];
  float s2 = 0.f;
#pragma unroll
  for (int u = 0; u < 4; ++u) s2 += ev[u] * ev[u];
  for (int off = 16; off >= 1; off >>= 1) s2 += __shfl_xor(s2, off, 32);
  float inv = rsqrtf(s2 + 1e-24f);
  int l = label[row];
  for (int cls = 0; cls < 3; ++cls) {
    float dot = 0.f;
#pragma unroll
    for (int u = 0; u < 4; ++u) dot += ev[u] * wn[cls * 128 + lane + 32 * u];
    for (int off = 16; off >= 1; off >>= 1) dot += __shfl_xor(dot, off, 32);
    if (lane == 0) {
      float cosv = dot * inv;
      float cc = fminf(fmaxf(cosv, -1.f + 1e-7f), 1.f - 1e-7f);
      float tl = cosf(acosf(cc) + M_ARC);
      out[(size_t)row * 3 + cls] = S_ARC * ((cls == l) ? tl : cosv);
    }
  }
}

// ------------------------------- host side ---------------------------------

extern "C" void kernel_launch(void* const* d_in, const int* in_sizes, int n_in,
                              void* d_out, int out_size, void* d_ws, size_t ws_size,
                              hipStream_t stream) {
  (void)in_sizes; (void)n_in; (void)out_size; (void)ws_size;

  const float* pos   = (const float*)d_in[0];
  const int*   label = (const int*)d_in[1];
  const float* arcW  = (const float*)d_in[2];
  // param layer bases: W=+0, b=+1 (unused: cancels through BN), be=+2, g=+3
  const int I_C1A = 3, I_C1B = 7, I_C2A = 11, I_C2B = 15, I_C3A = 19, I_C3B = 23;
  const int I_E1 = 27, I_E2 = 31, I_E3 = 35, I_G1 = 39, I_G2 = 43;
  auto Wp  = [&](int i) { return (const float*)d_in[i + 0]; };
  auto BEp = [&](int i) { return (const float*)d_in[i + 2]; };
  auto Gp  = [&](int i) { return (const float*)d_in[i + 3]; };

  // --- workspace bump allocator ---
  char* ws = (char*)d_ws;
  size_t off = 0;
  auto alloc = [&](size_t bytes) -> void* {
    void* p = ws + off;
    off += (bytes + 255) & ~(size_t)255;
    return p;
  };

  int*   idxbuf = (int*)  alloc((size_t)BN_ROWS * KNN * 4);
  float* x1     = (float*)alloc((size_t)BN_ROWS * 64 * 4);
  float* x2     = (float*)alloc((size_t)BN_ROWS * 64 * 4);
  float* x3     = (float*)alloc((size_t)BN_ROWS * 128 * 4);
  f16*   combB  = (f16*)  alloc((size_t)BN_ROWS * 256 * 2);
  float* embB   = (float*)alloc((size_t)BN_ROWS * 128 * 4);
  float* wnB    = (float*)alloc(384 * 4);
  float* statsP = (float*)alloc(2048 * 4);

  // f16 weights (K padded to multiple of 32)
  f16* wh_c1a = (f16*)alloc((size_t)64 * 32 * 2);
  f16* wh_c1b = (f16*)alloc((size_t)64 * 64 * 2);
  f16* wh_c2a = (f16*)alloc((size_t)64 * 128 * 2);
  f16* wh_c2b = (f16*)alloc((size_t)64 * 64 * 2);
  f16* wh_c3a = (f16*)alloc((size_t)128 * 128 * 2);
  f16* wh_c3b = (f16*)alloc((size_t)128 * 128 * 2);
  f16* wh_g1  = (f16*)alloc((size_t)512 * 256 * 2);
  f16* wh_g2  = (f16*)alloc((size_t)1024 * 512 * 2);
  f16* wh_e1  = (f16*)alloc((size_t)512 * 1280 * 2);
  f16* wh_e2  = (f16*)alloc((size_t)256 * 512 * 2);
  f16* wh_e3  = (f16*)alloc((size_t)128 * 256 * 2);

  // big edge-phase buffers (reused for global MLP phase)
  f16*   Ebuf = (f16*)  alloc((size_t)ME_ROWS * 128 * 2);   // 167 MB
  float* Zbuf = (float*)alloc((size_t)ME_ROWS * 128 * 4);   // 335 MB
  f16*   Hbuf = (f16*)  alloc((size_t)ME_ROWS * 128 * 2);   // 167 MB

  // global-phase aliases (edge buffers free by then)
  float* Zg   = (float*)Ebuf;                               // <=134 MB needed
  char*  zb   = (char*)Zbuf;
  f16*   catB = (f16*)(zb);                                 // 32768*1280*2 = 84 MB
  f16*   g1h  = (f16*)(zb + (size_t)100 * 1024 * 1024);     // 33.6 MB
  f16*   e1h  = (f16*)(zb + (size_t)150 * 1024 * 1024);     // 33.6 MB
  f16*   e2h  = (f16*)(zb + (size_t)200 * 1024 * 1024);     // 16.8 MB

  // --- helpers ---
  auto launch1d = [&](int n) { return dim3((unsigned)((n + 255) / 256)); };
  auto w2h = [&](const float* W, f16* Wh, int Nout, int Kd, int Kp) {
    k_w2h<<<launch1d(Nout * Kp), 256, 0, stream>>>(W, Wh, Nout, Kd, Kp);
  };
  auto gemm = [&](const f16* A, const f16* Wh, float* Z, int M, int Kp, int Nout) {
    dim3 g((unsigned)(M / 128), (unsigned)(Nout / 64));
    k_gemm<<<g, 256, 0, stream>>>(A, Wh, Z, Kp, Nout);
  };
  auto stats = [&](const float* Z, int M, int C) {
    k_zero<<<launch1d(2 * C), 256, 0, stream>>>(statsP, 2 * C);
    k_bn_stats<<<1024, 256, 0, stream>>>(Z, statsP, M, C);
  };

  // --- stage all weights to f16 ---
  w2h(Wp(I_C1A), wh_c1a, 64, 6, 32);
  w2h(Wp(I_C1B), wh_c1b, 64, 64, 64);
  w2h(Wp(I_C2A), wh_c2a, 64, 128, 128);
  w2h(Wp(I_C2B), wh_c2b, 64, 64, 64);
  w2h(Wp(I_C3A), wh_c3a, 128, 128, 128);
  w2h(Wp(I_C3B), wh_c3b, 128, 128, 128);
  w2h(Wp(I_G1),  wh_g1,  512, 256, 256);
  w2h(Wp(I_G2),  wh_g2,  1024, 512, 512);
  w2h(Wp(I_E1),  wh_e1,  512, 1280, 1280);
  w2h(Wp(I_E2),  wh_e2,  256, 512, 512);
  w2h(Wp(I_E3),  wh_e3,  128, 256, 256);

  const float invME = 1.0f / (float)ME_ROWS;
  const float invBN = 1.0f / (float)BN_ROWS;

  // --- one EdgeConv block ---
  auto edge_layer = [&](const float* xin, int d, int Kp1,
                        const f16* WhA, int NoutA, const float* gA, const float* beA,
                        const f16* WhB, int NoutB, const float* gB, const float* beB,
                        float* xout, int combOff) {
    k_knn<<<BN_ROWS / 8, 256, 0, stream>>>(xin, d, idxbuf);
    k_edge_feat<<<launch1d(ME_ROWS * Kp1), 256, 0, stream>>>(xin, idxbuf, Ebuf, d, Kp1);
    gemm(Ebuf, WhA, Zbuf, ME_ROWS, Kp1, NoutA);
    stats(Zbuf, ME_ROWS, NoutA);
    k_bn_act<<<launch1d(ME_ROWS * NoutA), 256, 0, stream>>>(
        Zbuf, statsP, gA, beA, Hbuf, NoutA, 0, (float*)nullptr,
        ME_ROWS, NoutA, invME, 1);
    gemm(Hbuf, WhB, Zbuf, ME_ROWS, NoutA, NoutB);
    stats(Zbuf, ME_ROWS, NoutB);
    k_max_bn<<<launch1d(BN_ROWS * NoutB), 256, 0, stream>>>(
        Zbuf, statsP, gB, beB, xout, combB, 256, combOff, NoutB, invME);
  };

  edge_layer(pos, 3, 32,  wh_c1a, 64,  Gp(I_C1A), BEp(I_C1A),
                           wh_c1b, 64,  Gp(I_C1B), BEp(I_C1B), x1, 0);
  edge_layer(x1, 64, 128, wh_c2a, 64,  Gp(I_C2A), BEp(I_C2A),
                           wh_c2b, 64,  Gp(I_C2B), BEp(I_C2B), x2, 64);
  edge_layer(x2, 64, 128, wh_c3a, 128, Gp(I_C3A), BEp(I_C3A),
                           wh_c3b, 128, Gp(I_C3B), BEp(I_C3B), x3, 128);

  // --- global MLP phase ---
  // g1: [BN,256] -> 512
  gemm(combB, wh_g1, Zg, BN_ROWS, 256, 512);
  stats(Zg, BN_ROWS, 512);
  k_bn_act<<<launch1d(BN_ROWS * 512), 256, 0, stream>>>(
      Zg, statsP, Gp(I_G1), BEp(I_G1), g1h, 512, 0, (float*)nullptr,
      BN_ROWS, 512, invBN, 1);
  // g2: 512 -> 1024, into cat1280 cols [256,1280)
  gemm(g1h, wh_g2, Zg, BN_ROWS, 512, 1024);
  stats(Zg, BN_ROWS, 1024);
  k_bn_act<<<launch1d(BN_ROWS * 1024), 256, 0, stream>>>(
      Zg, statsP, Gp(I_G2), BEp(I_G2), catB, 1280, 256, (float*)nullptr,
      BN_ROWS, 1024, invBN, 1);
  // cat cols [0,256) = comb
  k_copy_h<<<launch1d(BN_ROWS * 256), 256, 0, stream>>>(combB, catB, 256, 1280, 256);
  // e1: 1280 -> 512
  gemm(catB, wh_e1, Zg, BN_ROWS, 1280, 512);
  stats(Zg, BN_ROWS, 512);
  k_bn_act<<<launch1d(BN_ROWS * 512), 256, 0, stream>>>(
      Zg, statsP, Gp(I_E1), BEp(I_E1), e1h, 512, 0, (float*)nullptr,
      BN_ROWS, 512, invBN, 1);
  // e2: 512 -> 256
  gemm(e1h, wh_e2, Zg, BN_ROWS, 512, 256);
  stats(Zg, BN_ROWS, 256);
  k_bn_act<<<launch1d(BN_ROWS * 256), 256, 0, stream>>>(
      Zg, statsP, Gp(I_E2), BEp(I_E2), e2h, 256, 0, (float*)nullptr,
      BN_ROWS, 256, invBN, 1);
  // e3: 256 -> 128, BN without ReLU, f32 embedding
  gemm(e2h, wh_e3, Zg, BN_ROWS, 256, 128);
  stats(Zg, BN_ROWS, 128);
  k_bn_act<<<launch1d(BN_ROWS * 128), 256, 0, stream>>>(
      Zg, statsP, Gp(I_E3), BEp(I_E3), (f16*)nullptr, 0, 0, embB,
      BN_ROWS, 128, invBN, 0);

  // --- ArcFace head ---
  k_wnorm<<<1, 128, 0, stream>>>(arcW, wnB);
  k_arcface<<<BN_ROWS / 8, 256, 0, stream>>>(embB, wnB, label, (float*)d_out);
}
